// SpectralConv2d_78932908966338
// MI455X (gfx1250) — matrix operand order repaired and checked by reading, not compiled, not run on hardware
//
#include <hip/hip_runtime.h>

// ---------------------------------------------------------------------------
// SpectralConv2d (NUFFT FNO layer) for MI455X / gfx1250, f16 WMMA path.
//   B=8, C=64 in/out channels, N=16384 points, 32 modes per branch, scale 64.
// Three stages:
//   A: ftyT[b][br][j][c] = sum_n basis_fwd[j,n] * x[c,n]      (WMMA, split-K atomics)
//   B: Ainv[b][br][o][j] = mode-mixed weights (complex 64x64, sign-folded)
//   C: out[b][o][n] = 64 * sum_br sum_j Ainv[o,j]*basis_inv[j,n]  (WMMA)
// ---------------------------------------------------------------------------

typedef __attribute__((ext_vector_type(16))) _Float16 v16h;
typedef __attribute__((ext_vector_type(8)))  _Float16 v8h;
typedef __attribute__((ext_vector_type(8)))  float    v8f;

#define NB     8
#define NC     64
#define NPTS   16384
#define NMODE  32          // modes per branch; j-dim = 64 (re/im interleaved)
#define TWOPI  6.28318530717958647692f

#define CHUNK_A 512
#define PADA    8          // fwd-basis row stride = CHUNK_A + PADA halves
#define CHUNK_C 512
#define PADC    8          // inv-basis row stride = 64 + PADC halves

__device__ __forceinline__ v16h cat8(v8h lo, v8h hi) {
  return __builtin_shufflevector(lo, hi, 0, 1, 2, 3, 4, 5, 6, 7,
                                         8, 9, 10, 11, 12, 13, 14, 15);
}

__device__ __forceinline__ v16h load16f_h(const float* __restrict__ p) {
  const float4* q = (const float4*)p;   // 64B-aligned by construction
  v16h r;
#pragma unroll
  for (int i = 0; i < 4; ++i) {
    float4 f = q[i];
    r[4 * i + 0] = (_Float16)f.x;
    r[4 * i + 1] = (_Float16)f.y;
    r[4 * i + 2] = (_Float16)f.z;
    r[4 * i + 3] = (_Float16)f.w;
  }
  return r;
}

// ---------------------------------------------------------------------------
// Stage A: forward NUFFT.  grid = (NPTS/CHUNK_A, NB), block = 256 (8 waves).
// LDS: fwd basis [2 br][64 j][CHUNK_A+PADA] f16  (133,120 B, dynamic).
// D = A(basis, M=j modes) x B(xT, N=channels), K = n;  ftyT += D via atomics.
// ---------------------------------------------------------------------------
__global__ __launch_bounds__(256) void nufft_fwd(const float* __restrict__ x,
                                                 const float* __restrict__ x_in,
                                                 float* __restrict__ ftyT) {
  extern __shared__ _Float16 sm[];
  const int RS    = CHUNK_A + PADA;
  const int nbase = blockIdx.x * CHUNK_A;
  const int b     = blockIdx.y;
  const int t     = threadIdx.x;

  // Build e^{-i 2pi p k}, k=0..31, via complex recurrence (1 sincos per n,br).
#pragma unroll
  for (int r = 0; r < CHUNK_A / 256; ++r) {
    const int nl = t + 256 * r;
#pragma unroll
    for (int br = 0; br < 2; ++br) {
      const float p = x_in[(((size_t)b * NPTS) + nbase + nl) * 2 + br];
      float sn, cs;
      __sincosf(TWOPI * p, &sn, &cs);
      float cr = 1.0f, ci = 0.0f;
      _Float16* col = sm + (size_t)br * 64 * RS + nl;
#pragma unroll
      for (int k = 0; k < NMODE; ++k) {
        col[(2 * k) * RS]     = (_Float16)cr;   // +cos(k th)
        col[(2 * k + 1) * RS] = (_Float16)ci;   // -sin(k th)
        const float nr = cr * cs + ci * sn;     // *= exp(-i th)
        const float ni = ci * cs - cr * sn;
        cr = nr; ci = ni;
      }
    }
  }
  __syncthreads();

  const int wave = t >> 5, lane = t & 31;
  const int l16 = lane >> 4, lm = lane & 15;
  const int Mt  = wave >> 1;             // j-tile (0..3), shared by both c-tiles
  const int Nt0 = (wave & 1) * 2;        // c-tiles
  const int Nt1 = Nt0 + 1;

  v8f acc[2][2] = {};                    // [branch][c-tile]
  const _Float16* arow0 = sm + (size_t)(0 * 64 + Mt * 16 + lm) * RS;
  const _Float16* arow1 = sm + (size_t)(1 * 64 + Mt * 16 + lm) * RS;
  const float* xr0 = x + (((size_t)b * NC) + Nt0 * 16 + lm) * NPTS + nbase;
  const float* xr1 = x + (((size_t)b * NC) + Nt1 * 16 + lm) * NPTS + nbase;

  for (int ks = 0; ks < CHUNK_A / 32; ++ks) {
    // A operand: contiguous K order, per-lane 16 consecutive K (k = 16*l16 + s).
    const int na = ks * 32 + l16 * 16;
    v16h a0 = cat8(*(const v8h*)(arow0 + na), *(const v8h*)(arow0 + na + 8));
    v16h a1 = cat8(*(const v8h*)(arow1 + na), *(const v8h*)(arow1 + na + 8));
    // B operand: per-lane 16 consecutive K -> 16 consecutive f32 from x.
    const int nb = ks * 32 + l16 * 16;
    v16h b0 = load16f_h(xr0 + nb);
    v16h b1 = load16f_h(xr1 + nb);
    acc[0][0] = __builtin_amdgcn_wmma_f32_16x16x32_f16(false, a0, false, b0,
                                  (short)0, acc[0][0], false, false);
    acc[0][1] = __builtin_amdgcn_wmma_f32_16x16x32_f16(false, a0, false, b1,
                                  (short)0, acc[0][1], false, false);
    acc[1][0] = __builtin_amdgcn_wmma_f32_16x16x32_f16(false, a1, false, b0,
                                  (short)0, acc[1][0], false, false);
    acc[1][1] = __builtin_amdgcn_wmma_f32_16x16x32_f16(false, a1, false, b1,
                                  (short)0, acc[1][1], false, false);
  }

  // C/D layout: VGPR r -> (M = r + 8*l16, N = lm).  Split-K accumulate.
#pragma unroll
  for (int br = 0; br < 2; ++br) {
    float* fb = ftyT + (((size_t)b * 2 + br) * 64 + Mt * 16 + 8 * l16) * 64 + lm;
#pragma unroll
    for (int r = 0; r < 8; ++r) {
      unsafeAtomicAdd(fb + (size_t)r * 64 + Nt0 * 16, acc[br][0][r]);
      unsafeAtomicAdd(fb + (size_t)r * 64 + Nt1 * 16, acc[br][1][r]);
    }
  }
}

// ---------------------------------------------------------------------------
// Stage B: mode mixing.  grid = (2, NB), block = 256.
//   my[o,k] = sum_i fty[i,k] * W[i,o,k]   (complex)
//   Ainv[o][2k] = my_re, Ainv[o][2k+1] = -my_im   (f16, scale 64 deferred)
// ---------------------------------------------------------------------------
__global__ __launch_bounds__(256) void nufft_mix(const float* __restrict__ ftyT,
                                                 const float* __restrict__ w1,
                                                 const float* __restrict__ w2,
                                                 _Float16* __restrict__ Ainv) {
  const int br = blockIdx.x;
  const int b  = blockIdx.y;
  const float* W = (br == 0) ? w1 : w2;          // y-branch -> w1, x-branch -> w2
  const float* F = ftyT + ((size_t)b * 2 + br) * 4096;
  _Float16*    A = Ainv + ((size_t)b * 2 + br) * 4096;

  __shared__ float Fs[4096];
  for (int i = threadIdx.x; i < 4096; i += 256) Fs[i] = F[i];
  __syncthreads();

  for (int p = threadIdx.x; p < NC * NMODE; p += 256) {
    const int o = p & 63, k = p >> 6;
    float mr = 0.0f, mi = 0.0f;
#pragma unroll 4
    for (int i = 0; i < NC; ++i) {
      const float fr = Fs[(2 * k) * 64 + i];
      const float fi = Fs[(2 * k + 1) * 64 + i];
      const float wr = W[(((size_t)i * 64 + o) * NMODE + k) * 2 + 0];
      const float wi = W[(((size_t)i * 64 + o) * NMODE + k) * 2 + 1];
      mr += fr * wr - fi * wi;
      mi += fr * wi + fi * wr;
    }
    A[o * 64 + 2 * k]     = (_Float16)mr;
    A[o * 64 + 2 * k + 1] = (_Float16)(-mi);
  }
}

// ---------------------------------------------------------------------------
// Stage C: inverse NUFFT.  grid = (NPTS/CHUNK_C, NB), block = 256 (8 waves).
// LDS: inv basis [2 br][CHUNK_C n][64+PADC j] f16  (147,456 B, dynamic).
// out[b,o,n] = 64 * sum_br sum_j Ainv[o,j] * basis_inv[j,n]
// ---------------------------------------------------------------------------
__global__ __launch_bounds__(256) void nufft_inv(const float* __restrict__ x_out,
                                                 const _Float16* __restrict__ Ainv,
                                                 float* __restrict__ out) {
  extern __shared__ _Float16 sm[];
  const int RS    = 64 + PADC;
  const int nbase = blockIdx.x * CHUNK_C;
  const int b     = blockIdx.y;
  const int t     = threadIdx.x;

  // Build e^{+i 2pi q k}: j=2k -> cos, j=2k+1 -> +sin (sign folded into Ainv).
#pragma unroll
  for (int r = 0; r < CHUNK_C / 256; ++r) {
    const int nl = t + 256 * r;
#pragma unroll
    for (int br = 0; br < 2; ++br) {
      const float q = x_out[(((size_t)b * NPTS) + nbase + nl) * 2 + br];
      float sn, cs;
      __sincosf(TWOPI * q, &sn, &cs);
      float cr = 1.0f, ci = 0.0f;
      _Float16* row = sm + ((size_t)br * CHUNK_C + nl) * RS;
#pragma unroll
      for (int k = 0; k < NMODE; ++k) {
        row[2 * k]     = (_Float16)cr;
        row[2 * k + 1] = (_Float16)ci;
        const float nr = cr * cs - ci * sn;    // *= exp(+i th)
        const float ni = ci * cs + cr * sn;
        cr = nr; ci = ni;
      }
    }
  }
  __syncthreads();

  const int wave = t >> 5, lane = t & 31;
  const int l16 = lane >> 4, lm = lane & 15;
  const int Mt      = wave & 3;     // o-tile
  const int ntStart = wave >> 2;    // n-tile parity

  // Hoist the 4 A operands (2 branches x 2 j-steps) into registers.
  v16h Aop[2][2];
  const _Float16* Ab =
      Ainv + ((size_t)b * 2) * 4096 + (size_t)(Mt * 16 + lm) * 64 + l16 * 8;
#pragma unroll
  for (int br = 0; br < 2; ++br)
#pragma unroll
    for (int js = 0; js < 2; ++js) {
      const _Float16* ap = Ab + (size_t)br * 4096 + js * 32;
      Aop[br][js] = cat8(*(const v8h*)ap, *(const v8h*)(ap + 16));
    }

  for (int nt = ntStart; nt < CHUNK_C / 16; nt += 2) {
    v8f acc = {};
#pragma unroll
    for (int br = 0; br < 2; ++br)
#pragma unroll
      for (int js = 0; js < 2; ++js) {
        const _Float16* bp = sm + ((size_t)br * CHUNK_C + nt * 16 + lm) * RS +
                             js * 32 + l16 * 8;
        v16h bm = cat8(*(const v8h*)bp, *(const v8h*)(bp + 16));
        acc = __builtin_amdgcn_wmma_f32_16x16x32_f16(false, Aop[br][js], false,
                                     bm, (short)0, acc, false, false);
      }
    float* ob = out + (((size_t)b * NC) + Mt * 16 + 8 * l16) * NPTS + nbase +
                nt * 16 + lm;
#pragma unroll
    for (int r = 0; r < 8; ++r) ob[(size_t)r * NPTS] = 64.0f * acc[r];
  }
}

// ---------------------------------------------------------------------------
extern "C" void kernel_launch(void* const* d_in, const int* in_sizes, int n_in,
                              void* d_out, int out_size, void* d_ws,
                              size_t ws_size, hipStream_t stream) {
  const float* x     = (const float*)d_in[0];
  const float* x_in  = (const float*)d_in[1];
  const float* x_out = (const float*)d_in[2];
  const float* w1    = (const float*)d_in[3];
  const float* w2    = (const float*)d_in[4];
  float* out = (float*)d_out;

  float*    ftyT = (float*)d_ws;                                   // 256 KB
  _Float16* Ainv = (_Float16*)((char*)d_ws + (size_t)NB * 2 * 64 * 64 * 4);

  // Zero the split-K accumulation buffer (graph-capturable memset).
  hipMemsetAsync(ftyT, 0, (size_t)NB * 2 * 64 * 64 * sizeof(float), stream);

  dim3 gA(NPTS / CHUNK_A, NB);
  size_t ldsA = (size_t)2 * 64 * (CHUNK_A + PADA) * sizeof(_Float16);
  nufft_fwd<<<gA, 256, ldsA, stream>>>(x, x_in, ftyT);

  dim3 gB(2, NB);
  nufft_mix<<<gB, 256, 0, stream>>>(ftyT, w1, w2, Ainv);

  dim3 gC(NPTS / CHUNK_C, NB);
  size_t ldsC = (size_t)2 * CHUNK_C * (64 + PADC) * sizeof(_Float16);
  nufft_inv<<<gC, 256, ldsC, stream>>>(x_out, Ainv, out);
}
